// AdaptiveGaussianConvLayer_33071248179557
// MI455X (gfx1250) — matrix operand order repaired and checked by reading, not compiled, not run on hardware
//
#include <hip/hip_runtime.h>

typedef __attribute__((ext_vector_type(16))) _Float16 v16h;
typedef __attribute__((ext_vector_type(8)))  _Float16 v8h;
typedef __attribute__((ext_vector_type(8)))  float    v8f;

#if __has_builtin(__builtin_amdgcn_exp2f)
#define FAST_EXP2(x) __builtin_amdgcn_exp2f(x)
#else
#define FAST_EXP2(x) exp2f(x)
#endif

// Problem shape (fixed by the reference): B=4, N=4096, D=512.
constexpr int TILE_J = 128;   // j rows per block (8 waves x 16)
constexpr int TILE_D = 128;   // d cols per block (8 WMMA tiles per wave)
constexpr int TILE_K = 32;    // contraction (i) step = WMMA K
constexpr int KSTRIDE = 40;   // f16 per LDS d-row: 32 + 8 pad = 80B (16B aligned)
// sqrt(0.5 * log2(e)): fold the Gaussian constant into 1/sigma so the inner
// loop computes exp2(-(x*s)^2) with the negate as a free source modifier.
constexpr float SQRT_HALF_LOG2E = 0.84932180028801904272f;

struct SMem {
    _Float16 v[2][TILE_D * KSTRIDE]; // staged V tile, f16, transposed [d][k]
    float2   q[2][TILE_K];           // per-i: (i + mu_i, sqrt(ln2e/2)/sigma_i)
};

__device__ __forceinline__ void stage_tile(SMem& sm, int st, int kbase,
                                           const float* __restrict__ Vb,
                                           const float* __restrict__ mub,
                                           const float* __restrict__ sgb,
                                           int dbase, int tid, int N, int D) {
    // 256 threads: thread -> one d column (tid&127), one k half-row block (tid>>7).
    // Global reads are lane-consecutive in d (perfectly coalesced b32 loads);
    // LDS writes are two contiguous 16B f16 vectors per thread.
    const int dl = tid & (TILE_D - 1);
    const int kh = (tid >> 7) & 1;
    const float* vp = Vb + (size_t)(kbase + kh * 16) * D + (dbase + dl);
    v8h h0, h1;
#pragma unroll
    for (int k = 0; k < 8; ++k) h0[k] = (_Float16)vp[(size_t)k * D];
#pragma unroll
    for (int k = 0; k < 8; ++k) h1[k] = (_Float16)vp[(size_t)(k + 8) * D];
    _Float16* dst = &sm.v[st][dl * KSTRIDE + kh * 16];
    *(v8h*)(dst)     = h0;
    *(v8h*)(dst + 8) = h1;

    // Prefetch the K+2 tile (speculative; dropped if OOB translation fails).
    if (kbase + 2 * TILE_K < N)
        __builtin_prefetch(vp + (size_t)(2 * TILE_K) * D, 0, 0);

    if (tid < TILE_K) {
        const int i = kbase + tid;
        sm.q[st][tid] = make_float2((float)i + mub[i],
                                    __builtin_amdgcn_rcpf(sgb[i]) * SQRT_HALF_LOG2E);
    }
}

__global__ __launch_bounds__(256)
void gauss_conv_wmma(const float* __restrict__ V,
                     const float* __restrict__ sigma,
                     const float* __restrict__ mu,
                     float* __restrict__ out,
                     int N, int D) {
    __shared__ SMem sm;

    const int tid   = threadIdx.x;
    const int lane  = tid & 31;
    const int wv    = tid >> 5;           // wave id 0..7 (wave32)
    const int m     = lane & 15;          // A row within tile / B+C column
    const int half  = lane >> 4;          // lane half selects K sub-block
    const int bb    = blockIdx.z;
    const int jbase = blockIdx.x * TILE_J;
    const int dbase = blockIdx.y * TILE_D;

    const float* Vb  = V     + (size_t)bb * N * D;
    const float* mub = mu    + (size_t)bb * N;
    const float* sgb = sigma + (size_t)bb * N;

    const float jf    = (float)(jbase + wv * 16 + m); // this lane's j
    const int   ka    = half * 8;                     // A-frag K offset (per half)
    const int   koffb = half * 8;                     // B-frag K offset (per half)

    v8f acc[8];
#pragma unroll
    for (int t = 0; t < 8; ++t) acc[t] = (v8f){0.f,0.f,0.f,0.f,0.f,0.f,0.f,0.f};

    const int NK = N / TILE_K;
    stage_tile(sm, 0, 0, Vb, mub, sgb, dbase, tid, N, D);
    __syncthreads();

    for (int ks = 0; ks < NK; ++ks) {
        const int st = ks & 1;
        if (ks + 1 < NK)
            stage_tile(sm, st ^ 1, (ks + 1) * TILE_K, Vb, mub, sgb, dbase, tid, N, D);

        // Build A fragment (W^T tile) in registers, exact ISA 16-bit A 16x32 layout:
        //   e<8 : K = ka + e          (lanes 0-15: K 0..7,  lanes 16-31: K 8..15)
        //   e>=8: K = 16 + ka + (e-8) (lanes 0-15: K 16..23, lanes 16-31: K 24..31)
        // Per element: v_sub, v_mul, v_mul(neg mod), v_exp, v_cvt.
        v16h a;
#pragma unroll
        for (int e = 0; e < 16; ++e) {
            const int k = (e < 8) ? (ka + e) : (8 + ka + e);
            const float2 qs = sm.q[st][k];        // (i + mu_i, c/sigma_i)
            const float x = jf - qs.x;            // j - i - mu_i
            const float z = x * qs.y;
            a[e] = (_Float16)FAST_EXP2(-z * z);   // exp(-0.5*((j-i-mu)/sigma)^2)
        }

        // 8 WMMAs share one A fragment; B fragments from LDS (lane = column d,
        // K sequential -> two contiguous ds_load_b128 per fragment).
#pragma unroll
        for (int t = 0; t < 8; ++t) {
            const _Float16* bp = &sm.v[st][(t * 16 + m) * KSTRIDE + koffb];
            const v8h lo = *(const v8h*)bp;
            const v8h hi = *(const v8h*)(bp + 16);
            const v16h bf = __builtin_shufflevector(lo, hi,
                0,1,2,3,4,5,6,7,8,9,10,11,12,13,14,15);
            acc[t] = __builtin_amdgcn_wmma_f32_16x16x32_f16(
                false, a, false, bf, (short)0, acc[t], false, false);
        }
        __syncthreads();
    }

    // C/D layout: VGPR r, lanes 0-15 -> row r, lanes 16-31 -> row r+8; col = lane&15.
    float* outp = out + ((size_t)bb * N + jbase + wv * 16) * D + dbase;
    const int mh = half * 8;
#pragma unroll
    for (int t = 0; t < 8; ++t)
#pragma unroll
        for (int r = 0; r < 8; ++r)
            outp[(size_t)(mh + r) * D + t * 16 + m] = acc[t][r];
}

extern "C" void kernel_launch(void* const* d_in, const int* in_sizes, int n_in,
                              void* d_out, int out_size, void* d_ws, size_t ws_size,
                              hipStream_t stream) {
    const float* V     = (const float*)d_in[0];
    const float* sigma = (const float*)d_in[1];
    const float* mu    = (const float*)d_in[2];
    float* out = (float*)d_out;

    const int N  = 4096;                     // reference shape
    const int BN = in_sizes[1];              // B*N
    const int D  = in_sizes[0] / BN;         // 512
    const int B  = BN / N;                   // 4

    dim3 grid(N / TILE_J, D / TILE_D, B);    // 32 x 4 x 4 = 512 blocks
    dim3 block(256);
    gauss_conv_wmma<<<grid, block, 0, stream>>>(V, sigma, mu, out, N, D);
}
